// Decoder_46377056862572
// MI455X (gfx1250) — matrix operand order repaired and checked by reading, not compiled, not run on hardware
//
#include <hip/hip_runtime.h>
#include <hip/hip_bf16.h>

#define B_  128
#define P_  196
#define E_  2048
#define H_  512
#define V_  10000
#define T_  20
#define TS_ 19          // T-1 decode steps
#define KL_ 3072        // H + E + H  (emb | gated ctx | h)
#define VP_ 10112       // V padded to multiple of 128

typedef __attribute__((ext_vector_type(16))) __bf16 v16bf;
typedef __attribute__((ext_vector_type(8)))  __bf16 v8bf;
typedef __attribute__((ext_vector_type(8)))  float  v8f;
typedef __attribute__((ext_vector_type(4)))  float  v4f;

// Branch-free transcendental helpers: v_exp_f32 + v_rcp_f32, no exec-mask games.
// fast_tanh saturates to +/-1 correctly (exp->inf => rcp->0; exp->0 => 1-2= -1).
__device__ __forceinline__ float fast_tanh(float x) {
  float e = __expf(2.0f * x);
  return 1.0f - 2.0f * __builtin_amdgcn_rcpf(e + 1.0f);
}
__device__ __forceinline__ float fast_sig(float x) {
  return __builtin_amdgcn_rcpf(1.0f + __expf(-x));
}

// ---- WMMA fragment loaders (wave32 16x16x32 bf16 layouts per CDNA5 ISA 7.12.2) ----
__device__ __forceinline__ v16bf frag_bf16(const __bf16* p1, const __bf16* p2) {
  v8bf a = *(const v8bf*)p1;
  v8bf b = *(const v8bf*)p2;
  v16bf r;
#pragma unroll
  for (int i = 0; i < 8; ++i) { r[i] = a[i]; r[8 + i] = b[i]; }
  return r;
}
__device__ __forceinline__ v16bf load_frag(const __bf16* rowp, int k1, int k2) {
  return frag_bf16(rowp + k1, rowp + k2);
}
__device__ __forceinline__ v16bf load_frag(const float* rowp, int k1, int k2) {
  v4f a0 = *(const v4f*)(rowp + k1);
  v4f a1 = *(const v4f*)(rowp + k1 + 4);
  v4f b0 = *(const v4f*)(rowp + k2);
  v4f b1 = *(const v4f*)(rowp + k2 + 4);
  v16bf r;
#pragma unroll
  for (int i = 0; i < 4; ++i) {
    r[i]      = (__bf16)a0[i]; r[4 + i]  = (__bf16)a1[i];
    r[8 + i]  = (__bf16)b0[i]; r[12 + i] = (__bf16)b1[i];
  }
  return r;
}

// ---- Generic WMMA GEMM: C[M,N] = act(A[M,K] @ Bt[N,K]^T + bias) ----
// Contract: M % 32 == 0, K % 32 == 0, Npad % 128 == 0, bias != null.
// Block = 64 threads = 2 waves; each wave computes a 32(M) x 64(N) tile: 8 accumulators,
// 8 v_wmma_f32_16x16x32_bf16 per K=32 step.  ACT: 0=none 1=tanh 2=sigmoid.
template <typename AT, typename CT, int ACT>
__global__ __launch_bounds__(64)
void wmma_gemm(const AT* __restrict__ A, int lda,
               const __bf16* __restrict__ Bt, int ldk,
               const float* __restrict__ bias,
               CT* __restrict__ C, long ldc,
               int Nvalid, int K) {
  const int lane = threadIdx.x & 31;
  const int wave = threadIdx.x >> 5;
  const int hi   = lane >> 4;
  const int l16  = lane & 15;
  const int nbase = blockIdx.x * 128 + wave * 64;
  const int mbase = blockIdx.y * 32;

  v8f acc[2][4];
#pragma unroll
  for (int i = 0; i < 2; ++i)
#pragma unroll
    for (int j = 0; j < 4; ++j) acc[i][j] = (v8f){0.f,0.f,0.f,0.f,0.f,0.f,0.f,0.f};

  const AT* arow0 = A + (long)(mbase + l16) * lda;
  const AT* arow1 = A + (long)(mbase + 16 + l16) * lda;

  for (int k0 = 0; k0 < K; k0 += 32) {
    const int ka = k0 + hi * 8;          // A: two 8-elem chunks (k, k+16)
    v16bf a0 = load_frag(arow0, ka, ka + 16);
    v16bf a1 = load_frag(arow1, ka, ka + 16);
    const int kb = k0 + hi * 8;          // B: two 8-elem chunks (k, k+16)
    v16bf bfr[4];
#pragma unroll
    for (int j = 0; j < 4; ++j) {
      const __bf16* bp = Bt + (long)(nbase + j * 16 + l16) * ldk + kb;
      bfr[j] = frag_bf16(bp, bp + 16);
    }
#pragma unroll
    for (int j = 0; j < 4; ++j)
      acc[0][j] = __builtin_amdgcn_wmma_f32_16x16x32_bf16(
          false, a0, false, bfr[j], (short)0, acc[0][j], false, false);
#pragma unroll
    for (int j = 0; j < 4; ++j)
      acc[1][j] = __builtin_amdgcn_wmma_f32_16x16x32_bf16(
          false, a1, false, bfr[j], (short)0, acc[1][j], false, false);
  }

#pragma unroll
  for (int i = 0; i < 2; ++i) {
#pragma unroll
    for (int j = 0; j < 4; ++j) {
      const int col = nbase + j * 16 + l16;
      if (col >= Nvalid) continue;
      const float bb = bias[col];
      CT* cp = C + (long)(mbase + i * 16 + hi * 8) * ldc + col;
#pragma unroll
      for (int r = 0; r < 8; ++r) {
        float v = acc[i][j][r] + bb;
        if (ACT == 1) v = fast_tanh(v);
        else if (ACT == 2) v = fast_sig(v);
        cp[(long)r * ldc] = (CT)v;
      }
    }
  }
}

// ---- Weight convert + transpose:  dst[n, kofs+k] = bf16(src[k, n]), zero-pad n>=N ----
__global__ void convert_w(const float* __restrict__ src, __bf16* __restrict__ dst,
                          int K, int N, int Npad, int ldk, int kofs) {
  long idx = (long)blockIdx.x * 256 + threadIdx.x;
  long total = (long)Npad * K;
  if (idx >= total) return;
  int n = (int)(idx / K);
  int k = (int)(idx - (long)n * K);
  float v = (n < N) ? src[(long)k * N + n] : 0.0f;
  dst[(long)n * ldk + kofs + k] = (__bf16)v;
}

// ---- avg over patches:  avg[b,e] = mean_p img[b,p,e] ----
__global__ void avg_kernel(const float* __restrict__ img, float* __restrict__ avg) {
  int e = blockIdx.x * 256 + threadIdx.x;
  int b = blockIdx.y;
  const float* ib = img + (long)b * P_ * E_ + e;
  float s = 0.0f;
#pragma unroll 4
  for (int p = 0; p < P_; ++p) s += ib[(long)p * E_];
  avg[(long)b * E_ + e] = s * (1.0f / (float)P_);
}

// ---- attention energies: e[b,p] = sum_h tanh(Ws[b,p,h] + hU[b,h]) * v[h] + bv ----
__global__ __launch_bounds__(256)
void e_kernel(const __bf16* __restrict__ Ws, const float* __restrict__ hU,
              const float* __restrict__ vatt, const float* __restrict__ bv,
              float* __restrict__ e_out) {
  int lane = threadIdx.x & 31;
  int wid  = threadIdx.x >> 5;
  int bp   = blockIdx.x * 8 + wid;          // 3136 * 8 = 25088 rows exactly
  int b    = bp / P_;
  const __bf16* wr = Ws + (long)bp * H_ + lane * 16;
  const float*  hr = hU + (long)b * H_ + lane * 16;
  const float*  vr = vatt + lane * 16;
  v8bf w0 = *(const v8bf*)wr;
  v8bf w1 = *(const v8bf*)(wr + 8);
  float s = 0.0f;
#pragma unroll
  for (int i = 0; i < 8; ++i) s += fast_tanh((float)w0[i] + hr[i]) * vr[i];
#pragma unroll
  for (int i = 0; i < 8; ++i) s += fast_tanh((float)w1[i] + hr[8 + i]) * vr[8 + i];
#pragma unroll
  for (int off = 16; off > 0; off >>= 1) s += __shfl_xor(s, off, 32);
  if (lane == 0) e_out[bp] = s + bv[0];
}

// ---- softmax over P per batch row; also writes alphas output slice ----
__global__ __launch_bounds__(256)
void softmax_kernel(const float* __restrict__ e_in, float* __restrict__ alpha,
                    float* __restrict__ alphas_out, int t) {
  __shared__ float sm[256];
  int b = blockIdx.x, tid = threadIdx.x;
  float v = (tid < P_) ? e_in[(long)b * P_ + tid] : -1e30f;
  sm[tid] = v; __syncthreads();
  for (int s = 128; s > 0; s >>= 1) { if (tid < s) sm[tid] = fmaxf(sm[tid], sm[tid + s]); __syncthreads(); }
  float mx = sm[0]; __syncthreads();
  float ex = (tid < P_) ? __expf(v - mx) : 0.0f;
  sm[tid] = ex; __syncthreads();
  for (int s = 128; s > 0; s >>= 1) { if (tid < s) sm[tid] += sm[tid + s]; __syncthreads(); }
  float inv = __builtin_amdgcn_rcpf(sm[0]);
  if (tid < P_) {
    float a = ex * inv;
    alpha[(long)b * P_ + tid] = a;
    alphas_out[((long)b * TS_ + t) * P_ + tid] = a;
  }
}

// ---- context + gating, fused into bf16 lstm_in middle section ----
__global__ __launch_bounds__(256)
void context_kernel(const float* __restrict__ img, const float* __restrict__ alpha,
                    const float* __restrict__ gate, __bf16* __restrict__ lstm_in) {
  __shared__ float al[P_];
  int b = blockIdx.y;
  int e = blockIdx.x * 256 + threadIdx.x;
  if (threadIdx.x < P_) al[threadIdx.x] = alpha[(long)b * P_ + threadIdx.x];
  __syncthreads();
  const float* ib = img + (long)b * P_ * E_ + e;
  float s = 0.0f;
#pragma unroll 4
  for (int p = 0; p < P_; ++p) s += al[p] * ib[(long)p * E_];
  lstm_in[(long)b * KL_ + H_ + e] = (__bf16)(gate[(long)b * E_ + e] * s);
}

// ---- teacher-forced embedding gather into bf16 lstm_in head section ----
__global__ __launch_bounds__(512)
void build_head(const float* __restrict__ Wemb, const int* __restrict__ captions,
                __bf16* __restrict__ lstm_in, int t) {
  int b = blockIdx.x, tid = threadIdx.x;
  int cap = captions[(long)b * T_ + t];
  lstm_in[(long)b * KL_ + tid] = (__bf16)Wemb[(long)cap * H_ + tid];
}

// ---- LSTM elementwise update; writes new h (bf16) into lstm_in tail section ----
__global__ __launch_bounds__(256)
void lstm_update(const float* __restrict__ gates, float* __restrict__ c,
                 __bf16* __restrict__ lstm_in) {
  int idx = blockIdx.x * 256 + threadIdx.x;   // 128*512
  int b = idx >> 9, j = idx & 511;
  const float* g = gates + (long)b * (4 * H_);
  float i_t = fast_sig(g[j]);
  float f_t = fast_sig(g[H_ + j]);
  float g_t = fast_tanh(g[2 * H_ + j]);
  float o_t = fast_sig(g[3 * H_ + j]);
  float cn = f_t * c[idx] + i_t * g_t;
  c[idx] = cn;
  lstm_in[(long)b * KL_ + 2 * H_ + E_ + j] = (__bf16)(o_t * fast_tanh(cn));
}

// -------------------- host side --------------------
static inline char* carve(char*& p, size_t bytes) {
  char* r = p;
  p += (bytes + 255) & ~(size_t)255;
  return r;
}

template <typename AT, typename CT, int ACT>
static void launch_gemm(const AT* A, int lda, const __bf16* Bt, int ldk, const float* bias,
                        CT* C, long ldc, int M, int Nvalid, int Npad, int K, hipStream_t s) {
  dim3 grid(Npad / 128, M / 32);
  wmma_gemm<AT, CT, ACT><<<grid, dim3(64), 0, s>>>(A, lda, Bt, ldk, bias, C, ldc, Nvalid, K);
}

extern "C" void kernel_launch(void* const* d_in, const int* in_sizes, int n_in,
                              void* d_out, int out_size, void* d_ws, size_t ws_size,
                              hipStream_t stream) {
  const float* img   = (const float*)d_in[0];
  const int*   cap   = (const int*)d_in[1];
  const float* U_att = (const float*)d_in[2];  const float* bU  = (const float*)d_in[3];
  const float* W_att = (const float*)d_in[4];  const float* bW  = (const float*)d_in[5];
  const float* v_att = (const float*)d_in[6];  const float* bv  = (const float*)d_in[7];
  const float* Wih   = (const float*)d_in[8];  const float* bih = (const float*)d_in[9];
  const float* Wic   = (const float*)d_in[10]; const float* bic = (const float*)d_in[11];
  const float* Wfb   = (const float*)d_in[12]; const float* bfb = (const float*)d_in[13];
  const float* Wdo   = (const float*)d_in[14]; const float* bdo = (const float*)d_in[15];
  const float* Wemb  = (const float*)d_in[16];
  const float* Wl    = (const float*)d_in[17]; const float* Ul  = (const float*)d_in[18];
  const float* bl    = (const float*)d_in[19];

  float* preds      = (float*)d_out;                            // [B, TS, V]
  float* alphas_out = (float*)d_out + (long)B_ * TS_ * V_;      // [B, TS, P]

  char* wp = (char*)d_ws;
  __bf16* Uatt_t = (__bf16*)carve(wp, (size_t)H_ * H_ * 2);
  __bf16* Watt_t = (__bf16*)carve(wp, (size_t)H_ * E_ * 2);
  __bf16* Wfb_t  = (__bf16*)carve(wp, (size_t)E_ * H_ * 2);
  __bf16* Wih_t  = (__bf16*)carve(wp, (size_t)H_ * E_ * 2);
  __bf16* Wic_t  = (__bf16*)carve(wp, (size_t)H_ * E_ * 2);
  __bf16* Wdo_t  = (__bf16*)carve(wp, (size_t)VP_ * H_ * 2);
  __bf16* Wlc_t  = (__bf16*)carve(wp, (size_t)(4 * H_) * KL_ * 2);
  __bf16* Ws     = (__bf16*)carve(wp, (size_t)B_ * P_ * H_ * 2);
  __bf16* lstm_in = (__bf16*)carve(wp, (size_t)B_ * KL_ * 2);
  float* avg   = (float*)carve(wp, (size_t)B_ * E_ * 4);
  float* hU    = (float*)carve(wp, (size_t)B_ * H_ * 4);
  float* gate  = (float*)carve(wp, (size_t)B_ * E_ * 4);
  float* e_buf = (float*)carve(wp, (size_t)B_ * P_ * 4);
  float* alpha = (float*)carve(wp, (size_t)B_ * P_ * 4);
  float* gates = (float*)carve(wp, (size_t)B_ * 4 * H_ * 4);
  float* c_buf = (float*)carve(wp, (size_t)B_ * H_ * 4);
  if ((size_t)(wp - (char*)d_ws) > ws_size) return;  // workspace too small: bail

  auto conv = [&](const float* src, __bf16* dst, int K, int N, int Npad, int ldk, int kofs) {
    long total = (long)Npad * K;
    convert_w<<<dim3((unsigned)((total + 255) / 256)), dim3(256), 0, stream>>>(src, dst, K, N, Npad, ldk, kofs);
  };
  // Pre-transpose + bf16-cast all weights (Bt layout: [N, K])
  conv(U_att, Uatt_t, H_, H_, H_, H_, 0);
  conv(W_att, Watt_t, E_, H_, H_, E_, 0);
  conv(Wfb,   Wfb_t,  H_, E_, E_, H_, 0);
  conv(Wih,   Wih_t,  E_, H_, H_, E_, 0);
  conv(Wic,   Wic_t,  E_, H_, H_, E_, 0);
  conv(Wdo,   Wdo_t,  H_, V_, VP_, H_, 0);
  conv(Wl,    Wlc_t,  H_ + E_, 4 * H_, 4 * H_, KL_, 0);     // columns 0..2559
  conv(Ul,    Wlc_t,  H_, 4 * H_, 4 * H_, KL_, H_ + E_);    // columns 2560..3071

  // avg features, init h0 (bf16 directly into lstm_in tail), c0, and Ws precompute (bf16)
  avg_kernel<<<dim3(E_ / 256, B_), dim3(256), 0, stream>>>(img, avg);
  launch_gemm<float, __bf16, 1>(avg, E_, Wih_t, E_, bih, lstm_in + 2 * H_ + E_, KL_, B_, H_, H_, E_, stream);
  launch_gemm<float, float, 1>(avg, E_, Wic_t, E_, bic, c_buf, H_, B_, H_, H_, E_, stream);
  launch_gemm<float, __bf16, 0>(img, E_, Watt_t, E_, bW, Ws, H_, B_ * P_, H_, H_, E_, stream);

  const __bf16* h_bf = lstm_in + 2 * H_ + E_;  // h lives in lstm_in tail, lda = KL_

  for (int t = 0; t < TS_; ++t) {
    // attention pre-projection + f_beta gate (both consume current h)
    launch_gemm<__bf16, float, 0>(h_bf, KL_, Uatt_t, H_, bU, hU, H_, B_, H_, H_, H_, stream);
    launch_gemm<__bf16, float, 2>(h_bf, KL_, Wfb_t, H_, bfb, gate, E_, B_, E_, E_, H_, stream);
    e_kernel<<<dim3(B_ * P_ / 8), dim3(256), 0, stream>>>(Ws, hU, v_att, bv, e_buf);
    softmax_kernel<<<dim3(B_), dim3(256), 0, stream>>>(e_buf, alpha, alphas_out, t);
    context_kernel<<<dim3(E_ / 256, B_), dim3(256), 0, stream>>>(img, alpha, gate, lstm_in);
    build_head<<<dim3(B_), dim3(512), 0, stream>>>(Wemb, cap, lstm_in, t);
    // fused LSTM gates: [emb | g*ctx | h] @ [Wl; Ul]
    launch_gemm<__bf16, float, 0>(lstm_in, KL_, Wlc_t, KL_, bl, gates, 4 * H_, B_, 4 * H_, 4 * H_, KL_, stream);
    lstm_update<<<dim3(B_ * H_ / 256), dim3(256), 0, stream>>>(gates, c_buf, lstm_in);
    // deep output into preds[:, t, :]
    launch_gemm<__bf16, float, 0>(h_bf, KL_, Wdo_t, H_, bdo, preds + (long)t * V_, (long)TS_ * V_,
                                  B_, V_, VP_, H_, stream);
  }
}
